// LRCModel_51213190037770
// MI455X (gfx1250) — matrix operand order repaired and checked by reading, not compiled, not run on hardware
//
#include <hip/hip_runtime.h>
#include <hip/hip_bf16.h>
#include <math.h>

typedef __attribute__((ext_vector_type(16))) _Float16 v16h;
typedef __attribute__((ext_vector_type(8)))  _Float16 v8h;
typedef __attribute__((ext_vector_type(8)))  float    v8f;

union AFrag { v16h v; v8h h[2]; };

// ---------------------------------------------------------------------------
// Kernel 1: precompute separable layer-1 projections + swizzle W2/W3 into
// WMMA B-fragment-native f16 layout (per lane: 16 contiguous f16 = 32 B).
//   S1[s][c] = emb[s] @ W1[0:32]        (32 x 128)
//   T1[t][c] = emb[t] @ W1[32:64]       (32 x 128)
//   E1[e][c] = emb[u]@W1[64:96] + emb[v]@W1[96:128] + b1   (256 x 128)
//   W2f[kb][nb][lane][16], kb<4, nb<16 : B tile (K=kb*32.., N=nb*16..)
//   W3f[kb][nb][lane][16], kb<8, nb<8
// B-fragment lane layout chosen here, A matches (16-bit B 32x16): col = lane&15, K = (lane>>4)*16 + e
// ---------------------------------------------------------------------------
__global__ void k_pre(const float* __restrict__ emb, const int* __restrict__ edges,
                      const float* __restrict__ W1, const float* __restrict__ b1,
                      const float* __restrict__ W2, const float* __restrict__ W3,
                      float* __restrict__ S1, float* __restrict__ T1,
                      float* __restrict__ E1,
                      _Float16* __restrict__ W2f, _Float16* __restrict__ W3f)
{
    int tid = blockIdx.x * 256 + threadIdx.x;
    if (tid < 4096) {                                   // S1
        int s = tid >> 7, c = tid & 127;
        float v = 0.f;
        #pragma unroll 8
        for (int k = 0; k < 32; ++k) v += emb[s*32+k] * W1[k*128 + c];
        S1[tid] = v;
    } else if (tid < 8192) {                            // T1
        int i = tid - 4096; int t = i >> 7, c = i & 127;
        float v = 0.f;
        #pragma unroll 8
        for (int k = 0; k < 32; ++k) v += emb[t*32+k] * W1[(32+k)*128 + c];
        T1[i] = v;
    } else if (tid < 40960) {                           // E1
        int i = tid - 8192; int e = i >> 7, c = i & 127;
        int u = edges[2*e], vn = edges[2*e+1];
        float v = b1[c];
        #pragma unroll 8
        for (int k = 0; k < 32; ++k)
            v += emb[u*32+k]*W1[(64+k)*128 + c] + emb[vn*32+k]*W1[(96+k)*128 + c];
        E1[i] = v;
    } else if (tid < 73728) {                           // W2f swizzle (128x256)
        int i = tid - 40960;
        int e2 = i & 15, l = (i >> 4) & 31, nb = (i >> 9) & 15, kb = i >> 13;
        int k = kb*32 + (l >> 4)*16 + e2;
        int n = nb*16 + (l & 15);
        W2f[i] = (_Float16)W2[k*256 + n];
    } else if (tid < 106496) {                          // W3f swizzle (256x128)
        int i = tid - 73728;
        int e2 = i & 15, l = (i >> 4) & 31, nb = (i >> 9) & 7, kb = i >> 12;
        int k = kb*32 + (l >> 4)*16 + e2;
        int n = nb*16 + (l & 15);
        W3f[i] = (_Float16)W3[k*128 + n];
    }
}

// ---------------------------------------------------------------------------
// Kernel 2: main MLP. 1024 workgroups (one per (s,t)), 256 threads = 8 waves.
// Each wave owns 32 edge-rows = two 16-row M-tiles; private 8KB LDS region.
// Layer2: (16x128)@(128x256) via 16 nb x 4 kb WMMA; Layer3: 8 nb x 8 kb WMMA;
// Layer4: per-lane dot with W4 + shfl_xor reduction over the 16-lane N group.
// ---------------------------------------------------------------------------
__global__ void __launch_bounds__(256) k_mlp(
    const float* __restrict__ S1, const float* __restrict__ T1,
    const float* __restrict__ E1,
    const _Float16* __restrict__ W2f, const _Float16* __restrict__ W3f,
    const float* __restrict__ b2, const float* __restrict__ b3,
    const float* __restrict__ W4, const float* __restrict__ b4,
    float* __restrict__ preds)
{
    __shared__ _Float16 hbuf[8][16][256];   // 64 KB: per-wave 16x256 staging
    const int p  = blockIdx.x;
    const int s  = p >> 5, t = p & 31;
    const int w  = threadIdx.x >> 5;        // wave id 0..7
    const int l  = threadIdx.x & 31;        // lane 0..31
    const int hi = l >> 4;                  // K/M half select
    const int lo = l & 15;
    const float b4v = b4[0];
    const int arow = lo;                    // A-frag: row = lane&15
    const int kcol = hi * 16;               // A-frag: K chunk base per half

    for (int mt = 0; mt < 2; ++mt) {
        const int ebase = w*32 + mt*16;     // first edge row of this M-tile

        // ---- stage h1 = relu(S1[s] + T1[t] + E1[e]) as f16, 16x128 ----
        for (int i = 0; i < 64; ++i) {
            int f = i*32 + l;
            int r = f >> 7, c = f & 127;
            float v = S1[s*128 + c] + T1[t*128 + c] + E1[(ebase + r)*128 + c];
            hbuf[w][r][c] = (_Float16)fmaxf(v, 0.f);
        }
        __syncthreads();

        // ---- load layer-2 A fragments (16x32 f16 each, K = kb*32) ----
        AFrag a1[4];
        #pragma unroll
        for (int kb = 0; kb < 4; ++kb) {
            a1[kb].h[0] = *(const v8h*)&hbuf[w][arow][kb*32 + kcol];
            a1[kb].h[1] = *(const v8h*)&hbuf[w][arow][kb*32 + kcol + 8];
        }
        __syncthreads();

        // ---- layer 2: h2 = relu(h1 @ W2 + b2), written back as f16 16x256 ----
        for (int nb = 0; nb < 16; ++nb) {
            v8f cacc = {};
            #pragma unroll
            for (int kb = 0; kb < 4; ++kb) {
                v16h b = *(const v16h*)(W2f + ((kb*16 + nb)*32 + l)*16);
                cacc = __builtin_amdgcn_wmma_f32_16x16x32_f16(
                        false, a1[kb].v, false, b, (short)0, cacc, false, false);
            }
            int col = nb*16 + lo;
            float bias = b2[col];
            #pragma unroll
            for (int r = 0; r < 8; ++r)
                hbuf[w][r + 8*hi][col] = (_Float16)fmaxf(cacc[r] + bias, 0.f);
        }
        __syncthreads();

        // ---- load layer-3 A fragments (K = 256 -> 8 blocks) ----
        AFrag a2[8];
        #pragma unroll
        for (int kb = 0; kb < 8; ++kb) {
            a2[kb].h[0] = *(const v8h*)&hbuf[w][arow][kb*32 + kcol];
            a2[kb].h[1] = *(const v8h*)&hbuf[w][arow][kb*32 + kcol + 8];
        }

        // ---- layer 3 + layer 4 fused ----
        float pd[8] = {0,0,0,0,0,0,0,0};
        for (int nb = 0; nb < 8; ++nb) {
            v8f cacc = {};
            #pragma unroll
            for (int kb = 0; kb < 8; ++kb) {
                v16h b = *(const v16h*)(W3f + ((kb*8 + nb)*32 + l)*16);
                cacc = __builtin_amdgcn_wmma_f32_16x16x32_f16(
                        false, a2[kb].v, false, b, (short)0, cacc, false, false);
            }
            int col = nb*16 + lo;
            float bias = b3[col];
            float w4v  = W4[col];
            #pragma unroll
            for (int r = 0; r < 8; ++r)
                pd[r] += fmaxf(cacc[r] + bias, 0.f) * w4v;
        }

        // reduce over the 16 lanes of each N-group (rows r+8*hi)
        #pragma unroll
        for (int r = 0; r < 8; ++r) {
            for (int m = 1; m < 16; m <<= 1)
                pd[r] += __shfl_xor(pd[r], m, 32);
        }
        if (lo == 0) {
            #pragma unroll
            for (int r = 0; r < 8; ++r) {
                int e = ebase + r + 8*hi;
                preds[p*256 + e] = pd[r] + b4v;
            }
        }
        __syncthreads();
    }
}

// ---------------------------------------------------------------------------
// Kernel 3: per-pair routing matrix + 50-step power iteration (one wave per
// pair). Lane v holds x[v]; row u of A owned by lane u (no atomics -> fully
// deterministic). Writes delta[p][v] (0 when s==t).
// ---------------------------------------------------------------------------
__global__ void k_power(const int* __restrict__ edges,
                        const float* __restrict__ preds,
                        float* __restrict__ delta)
{
    __shared__ float A[32*32];
    const int p = blockIdx.x, s = p >> 5, t = p & 31;
    const int l = threadIdx.x;

    for (int i = 0; i < 32; ++i) A[i*32 + l] = 0.f;
    __syncthreads();

    // lane l owns row u == l : serial, order-fixed accumulation (dup edges ok)
    for (int e = 0; e < 256; ++e) {
        int u = edges[2*e];
        if (u == l) A[l*32 + edges[2*e+1]] += preds[p*256 + e];
    }
    if (l == s) A[s*32 + s] += 1.f;         // ones at (s,s), row s owned by lane s
    __syncthreads();

    float x = 1.f / 32.f;
    for (int it = 0; it < 50; ++it) {
        float y = 0.f;
        #pragma unroll 8
        for (int u = 0; u < 32; ++u)
            y += __shfl(x, u, 32) * A[u*32 + l];
        float n2 = y * y;
        for (int m = 1; m < 32; m <<= 1) n2 += __shfl_xor(n2, m, 32);
        x = y / (sqrtf(n2) + 1e-9f);
    }
    float xs = __shfl(x, s, 32);
    delta[p*32 + l] = (s != t) ? (x / (xs + 1e-9f)) : 0.f;
}

// ---------------------------------------------------------------------------
// Kernel 4: rbc[v] = sum_p delta[p][v], normalized to sum 1. Fixed-order sum.
// ---------------------------------------------------------------------------
__global__ void k_final(const float* __restrict__ delta, float* __restrict__ out)
{
    const int l = threadIdx.x;
    float a = 0.f;
    for (int pp = 0; pp < 1024; ++pp) a += delta[pp*32 + l];
    float tot = a;
    for (int m = 1; m < 32; m <<= 1) tot += __shfl_xor(tot, m, 32);
    out[l] = a / tot;
}

// ---------------------------------------------------------------------------
extern "C" void kernel_launch(void* const* d_in, const int* in_sizes, int n_in,
                              void* d_out, int out_size, void* d_ws, size_t ws_size,
                              hipStream_t stream)
{
    (void)in_sizes; (void)n_in; (void)out_size; (void)ws_size;
    const float* emb   = (const float*)d_in[1];
    const int*   edges = (const int*)  d_in[2];
    const float* W1    = (const float*)d_in[3];
    const float* b1    = (const float*)d_in[4];
    const float* W2    = (const float*)d_in[5];
    const float* b2    = (const float*)d_in[6];
    const float* W3    = (const float*)d_in[7];
    const float* b3    = (const float*)d_in[8];
    const float* W4    = (const float*)d_in[9];
    const float* b4    = (const float*)d_in[10];

    char* ws = (char*)d_ws;
    float*    S1    = (float*)   (ws + 0);        //  16 KB
    float*    T1    = (float*)   (ws + 16384);    //  16 KB
    float*    E1    = (float*)   (ws + 32768);    // 128 KB
    _Float16* W2f   = (_Float16*)(ws + 163840);   //  64 KB
    _Float16* W3f   = (_Float16*)(ws + 229376);   //  64 KB
    float*    preds = (float*)   (ws + 294912);   //   1 MB
    float*    delta = (float*)   (ws + 1343488);  // 128 KB
    float*    out   = (float*)d_out;

    k_pre  <<<416, 256, 0, stream>>>(emb, edges, W1, b1, W2, W3, S1, T1, E1, W2f, W3f);
    k_mlp  <<<1024, 256, 0, stream>>>(S1, T1, E1, W2f, W3f, b2, b3, W4, b4, preds);
    k_power<<<1024, 32, 0, stream>>>(edges, preds, delta);
    k_final<<<1, 32, 0, stream>>>(delta, out);
}
